// SEB_59201829208446
// MI455X (gfx1250) — matrix operand order repaired and checked by reading, not compiled, not run on hardware
//
#include <hip/hip_runtime.h>
#include <hip/hip_bf16.h>

// ---------------------------------------------------------------------------
// Problem constants
// ---------------------------------------------------------------------------
#define B_  32
#define C_  2048
#define D_  256
#define M_  784          // 28*28
#define XELEMS ((size_t)B_ * C_ * M_)     // 51,380,224
#define ZELEMS ((size_t)B_ * D_ * M_)     // 6,422,528
#define MATELEMS ((size_t)B_ * D_ * D_)   // 2,097,152
#define TRIU 32896                        // 256*257/2

typedef __attribute__((ext_vector_type(16))) __bf16 v16bf;
typedef __attribute__((ext_vector_type(8)))  float  v8f;
typedef __attribute__((ext_vector_type(4)))  unsigned int v4u;
typedef __attribute__((ext_vector_type(8)))  int v8i;
typedef __attribute__((ext_vector_type(4)))  int v4i;

union BF16Frag { v16bf v; unsigned short u[16]; uint4 q[2]; };

#define LDS_OFF(p) ((unsigned)(unsigned long long)(uintptr_t)(const void*)(p))

__device__ __forceinline__ unsigned short f2bf(float f) {
  unsigned int x = __float_as_uint(f);
  unsigned int r = x + 0x7FFFu + ((x >> 16) & 1u);   // round-to-nearest-even
  return (unsigned short)(r >> 16);
}

// ---------------------------------------------------------------------------
// Tensor Data Mover: 2-D tile of bf16 (2-byte) elements, global -> LDS.
// D# per CDNA5 ISA 8.3/8.4. tensor_d0 < tile_d0 => TDM zero-fills the
// overhang (hardware OOB), used for the 784 K/M boundaries.
// ---------------------------------------------------------------------------
__device__ __forceinline__ void tdm_load_2d(unsigned lds_off, const void* gaddr,
                                            unsigned tensor_d0, unsigned tile_d0,
                                            unsigned tile_d1, unsigned stride_elems) {
  unsigned long long ga = (unsigned long long)gaddr;
  v4u g0;
  g0.x = 1u;                                                // count=1 (user D#)
  g0.y = lds_off;                                           // LDS byte address
  g0.z = (unsigned)(ga & 0xFFFFFFFFu);                      // global_addr lo
  g0.w = (unsigned)((ga >> 32) & 0x01FFFFFFu) | (2u << 30); // addr hi | type=2
  v8i g1;
  g1[0] = (int)0x00010000u;                                 // data_size = 2 bytes
  g1[1] = (int)((tensor_d0 & 0xFFFFu) << 16);               // tensor_dim0 lo16
  g1[2] = (int)(((tensor_d0 >> 16) & 0xFFFFu) | ((tile_d1 & 0xFFFFu) << 16)); // tensor_dim1=tile_d1
  g1[3] = (int)(((tile_d1 >> 16) & 0xFFFFu) | ((tile_d0 & 0xFFFFu) << 16));   // tile_dim0
  g1[4] = (int)(tile_d1 & 0xFFFFu);                         // tile_dim1 (tile_dim2=0)
  g1[5] = (int)stride_elems;                                // tensor_dim0_stride
  g1[6] = 0;
  g1[7] = 0;
  v4i z4 = {0, 0, 0, 0};
#if __clang_major__ >= 23
  v8i z8 = {0, 0, 0, 0, 0, 0, 0, 0};
  __builtin_amdgcn_tensor_load_to_lds(g0, g1, z4, z4, z8, 0);
#else
  __builtin_amdgcn_tensor_load_to_lds(g0, g1, z4, z4, 0);
#endif
}

// ---------------------------------------------------------------------------
// 0a) conv_w fp32 -> bf16
// ---------------------------------------------------------------------------
__global__ __launch_bounds__(256) void cvt_w_kernel(const float* __restrict__ cw,
                                                    unsigned short* __restrict__ Wb) {
  int idx = blockIdx.x * 256 + threadIdx.x;
  if (idx < D_ * C_) Wb[idx] = f2bf(cw[idx]);
}

// 0b) x fp32 -> bf16 (one bandwidth pass; removes all cvt from conv loop)
__global__ __launch_bounds__(256) void cvt_x_kernel(const float* __restrict__ x,
                                                    unsigned short* __restrict__ xbf) {
  const size_t i4 = ((size_t)blockIdx.x * 256 + threadIdx.x) * 4;   // XELEMS % 4 == 0
  const float4 v = *(const float4*)(x + i4);
  uint2 o;
  o.x = (unsigned)f2bf(v.x) | ((unsigned)f2bf(v.y) << 16);
  o.y = (unsigned)f2bf(v.z) | ((unsigned)f2bf(v.w) << 16);
  *(uint2*)(xbf + i4) = o;
}

// ---------------------------------------------------------------------------
// 1) 1x1 conv as GEMM:  Z[b] (256x784) = W (256x2048) @ X[b] (2048x784)
//    Workgroup: 128x64 output, 8 waves, 2x2 tiles. W panel (128x32) and X
//    panel (32x64) double-buffered in LDS via TDM; TDM OOB zero-fill masks
//    the 784-column overhang. Inner loop: b128 W reads + u16 X gathers + 4 WMMA, no VALU.
// ---------------------------------------------------------------------------
__global__ __launch_bounds__(256) void conv_gemm_kernel(const unsigned short* __restrict__ Wb,
                                                        const unsigned short* __restrict__ xbf,
                                                        float* __restrict__ z) {
  __shared__ unsigned short ldsA[2][128 * 32];  // 2 x 8 KB: [row][k]
  __shared__ unsigned short ldsX[2][32 * 64];   // 2 x 4 KB: [k][m]
  const int wave = threadIdx.x >> 5;
  const int lane = threadIdx.x & 31;
  const int b    = blockIdx.z;
  const int row0 = blockIdx.y * 128;            // 2 row blocks
  const int col0 = blockIdx.x * 64;             // 13 col blocks (last partial)
  const int wr = wave >> 1, wc = wave & 1;      // 4 x 2 wave grid of 32x32
  const int r = lane & 15, half = lane >> 4;

  const unsigned short* xb = xbf + (size_t)b * C_ * M_;
  const unsigned short* wrow = Wb + (size_t)row0 * C_;
  const unsigned mrem = (unsigned)(M_ - col0);  // >= 16; TDM zero-fills beyond
  const unsigned offA[2] = { LDS_OFF(&ldsA[0][0]), LDS_OFF(&ldsA[1][0]) };
  const unsigned offX[2] = { LDS_OFF(&ldsX[0][0]), LDS_OFF(&ldsX[1][0]) };
  const int ra0 = wr * 32 + r, ra1 = ra0 + 16;

  if (wave == 0) {
    tdm_load_2d(offA[0], wrow, 32, 32, 128, C_);
    tdm_load_2d(offX[0], xb + col0, mrem, 64, 32, M_);
  }

  v8f acc00 = {}, acc01 = {}, acc10 = {}, acc11 = {};
  for (int s = 0; s < 64; ++s) {
    if (wave == 0) {
      if (s < 63) {
        const int kn = s * 32 + 32;
        const int nb = (s + 1) & 1;
        tdm_load_2d(offA[nb], wrow + kn, 32, 32, 128, C_);
        tdm_load_2d(offX[nb], xb + (size_t)kn * M_ + col0, mrem, 64, 32, M_);
        __builtin_amdgcn_s_wait_tensorcnt(2);   // step-s panels landed
      } else {
        __builtin_amdgcn_s_wait_tensorcnt(0);
      }
    }
    __syncthreads();                            // panels visible to all waves
    const unsigned short* pA = ldsA[s & 1];
    const unsigned short* pX = ldsX[s & 1];

    BF16Frag a0, a1, b0, b1;
    a0.q[0] = *(const uint4*)(pA + ra0 * 32 + half * 8);
    a0.q[1] = *(const uint4*)(pA + ra0 * 32 + 16 + half * 8);
    a1.q[0] = *(const uint4*)(pA + ra1 * 32 + half * 8);
    a1.q[1] = *(const uint4*)(pA + ra1 * 32 + 16 + half * 8);
#pragma unroll
    for (int e = 0; e < 16; ++e) {
      const int kb = (e < 8) ? half * 8 + e : 16 + half * 8 + (e - 8);
      b0.u[e] = pX[kb * 64 + wc * 32 + r];
      b1.u[e] = pX[kb * 64 + wc * 32 + 16 + r];
    }
    acc00 = __builtin_amdgcn_wmma_f32_16x16x32_bf16(false, a0.v, false, b0.v, (short)0, acc00, false, false);
    acc01 = __builtin_amdgcn_wmma_f32_16x16x32_bf16(false, a0.v, false, b1.v, (short)0, acc01, false, false);
    acc10 = __builtin_amdgcn_wmma_f32_16x16x32_bf16(false, a1.v, false, b0.v, (short)0, acc10, false, false);
    acc11 = __builtin_amdgcn_wmma_f32_16x16x32_bf16(false, a1.v, false, b1.v, (short)0, acc11, false, false);
    __syncthreads();                            // panels fully consumed
  }

  // epilogue: column-group validity is wave-uniform (784 = 49*16)
  float* zt = z + (size_t)b * D_ * M_;
  const int cb0 = col0 + wc * 32;
  const int cb1 = col0 + wc * 32 + 16;
#pragma unroll
  for (int v = 0; v < 8; ++v) {
    const int r0w = row0 + wr * 32 + v + 8 * half;
    const int r1w = r0w + 16;
    if (cb0 < M_) {
      zt[(size_t)r0w * M_ + cb0 + r] = acc00[v];
      zt[(size_t)r1w * M_ + cb0 + r] = acc10[v];
    }
    if (cb1 < M_) {
      zt[(size_t)r0w * M_ + cb1 + r] = acc01[v];
      zt[(size_t)r1w * M_ + cb1 + r] = acc11[v];
    }
  }
}

// ---------------------------------------------------------------------------
// 2) BN statistics per channel over (B, H, W)
// ---------------------------------------------------------------------------
__global__ __launch_bounds__(256) void bn_stats_kernel(const float* __restrict__ z,
                                                       float* __restrict__ mu,
                                                       float* __restrict__ istd) {
  const int ch = blockIdx.x;
  float s = 0.f, s2 = 0.f;
  for (int i = threadIdx.x; i < B_ * M_; i += 256) {
    const int b = i / M_, m = i - b * M_;
    const float v = z[(size_t)b * D_ * M_ + (size_t)ch * M_ + m];
    s += v; s2 += v * v;
  }
  __shared__ float sh[256], sh2[256];
  sh[threadIdx.x] = s; sh2[threadIdx.x] = s2;
  __syncthreads();
  for (int st = 128; st > 0; st >>= 1) {
    if (threadIdx.x < st) { sh[threadIdx.x] += sh[threadIdx.x + st];
                            sh2[threadIdx.x] += sh2[threadIdx.x + st]; }
    __syncthreads();
  }
  if (threadIdx.x == 0) {
    const float inv_n = 1.0f / (float)(B_ * M_);
    const float mean = sh[0] * inv_n;
    const float var  = sh2[0] * inv_n - mean * mean;
    mu[ch] = mean;
    istd[ch] = rsqrtf(var + 1e-5f);
  }
}

// ---------------------------------------------------------------------------
// 3) BN + ReLU -> bf16, plus per-(b,ch) spatial mean for cov centering
// ---------------------------------------------------------------------------
__global__ __launch_bounds__(256) void bn_relu_kernel(const float* __restrict__ z,
    const float* __restrict__ mu, const float* __restrict__ istd,
    const float* __restrict__ gamma, const float* __restrict__ beta,
    unsigned short* __restrict__ zb, float* __restrict__ rowmean) {
  const int b = blockIdx.x >> 8;
  const int ch = blockIdx.x & 255;
  const float m = mu[ch], is = istd[ch], g = gamma[ch], be = beta[ch];
  const float* src = z  + (size_t)b * D_ * M_ + (size_t)ch * M_;
  unsigned short* dst = zb + (size_t)b * D_ * M_ + (size_t)ch * M_;
  float s = 0.f;
  for (int i = threadIdx.x; i < M_; i += 256) {
    float v = (src[i] - m) * is * g + be;
    v = fmaxf(v, 0.f);
    s += v;
    dst[i] = f2bf(v);
  }
  __shared__ float sh[256];
  sh[threadIdx.x] = s; __syncthreads();
  for (int st = 128; st > 0; st >>= 1) {
    if (threadIdx.x < st) sh[threadIdx.x] += sh[threadIdx.x + st];
    __syncthreads();
  }
  if (threadIdx.x == 0) rowmean[b * D_ + ch] = sh[0] * (1.0f / (float)M_);
}

// ---------------------------------------------------------------------------
// 4) Covpool:  cov[b] = (1/M) Zb Zb^T - rm rm^T.  Same TDM pipeline; both
//    panels are rows of Zb (NT form, K contiguous). K=784 tail is handled by
//    TDM tensor_dim0 zero-fill: 25 branch-free K-steps.
// ---------------------------------------------------------------------------
__global__ __launch_bounds__(256) void cov_gemm_kernel(const unsigned short* __restrict__ zb,
                                                       const float* __restrict__ rowmean,
                                                       float* __restrict__ cov) {
  __shared__ unsigned short ldsA[2][128 * 32];  // rows row0..row0+127, [row][k]
  __shared__ unsigned short ldsB[2][64 * 32];   // rows col0..col0+63,  [row][k]
  const int b = blockIdx.y;
  const int wave = threadIdx.x >> 5;
  const int lane = threadIdx.x & 31;
  const int row0 = (blockIdx.x >> 2) * 128;
  const int col0 = (blockIdx.x & 3) * 64;
  const int wr = wave >> 1, wc = wave & 1;
  const int r = lane & 15, half = lane >> 4;

  const unsigned short* zbb = zb + (size_t)b * D_ * M_;
  const unsigned offA[2] = { LDS_OFF(&ldsA[0][0]), LDS_OFF(&ldsA[1][0]) };
  const unsigned offB[2] = { LDS_OFF(&ldsB[0][0]), LDS_OFF(&ldsB[1][0]) };
  const int ra0 = wr * 32 + r, ra1 = ra0 + 16;
  const int cb0 = wc * 32 + r, cb1 = cb0 + 16;

  if (wave == 0) {
    tdm_load_2d(offA[0], zbb + (size_t)row0 * M_, M_, 32, 128, M_);
    tdm_load_2d(offB[0], zbb + (size_t)col0 * M_, M_, 32, 64, M_);
  }

  v8f acc00 = {}, acc01 = {}, acc10 = {}, acc11 = {};
  for (int s = 0; s < 25; ++s) {                // 25*32 >= 784, tail zero-filled
    if (wave == 0) {
      if (s < 24) {
        const int kn = s * 32 + 32;
        const int nb = (s + 1) & 1;
        const unsigned rem = (unsigned)(M_ - kn);  // TDM zero-fills k >= 784
        tdm_load_2d(offA[nb], zbb + (size_t)row0 * M_ + kn, rem, 32, 128, M_);
        tdm_load_2d(offB[nb], zbb + (size_t)col0 * M_ + kn, rem, 32, 64, M_);
        __builtin_amdgcn_s_wait_tensorcnt(2);
      } else {
        __builtin_amdgcn_s_wait_tensorcnt(0);
      }
    }
    __syncthreads();
    const unsigned short* pA = ldsA[s & 1];
    const unsigned short* pB = ldsB[s & 1];

    BF16Frag a0, a1, b0, b1;
    a0.q[0] = *(const uint4*)(pA + ra0 * 32 + half * 8);
    a0.q[1] = *(const uint4*)(pA + ra0 * 32 + 16 + half * 8);
    a1.q[0] = *(const uint4*)(pA + ra1 * 32 + half * 8);
    a1.q[1] = *(const uint4*)(pA + ra1 * 32 + 16 + half * 8);
    b0.q[0] = *(const uint4*)(pB + cb0 * 32 + half * 8);
    b0.q[1] = *(const uint4*)(pB + cb0 * 32 + 16 + half * 8);
    b1.q[0] = *(const uint4*)(pB + cb1 * 32 + half * 8);
    b1.q[1] = *(const uint4*)(pB + cb1 * 32 + 16 + half * 8);
    acc00 = __builtin_amdgcn_wmma_f32_16x16x32_bf16(false, a0.v, false, b0.v, (short)0, acc00, false, false);
    acc01 = __builtin_amdgcn_wmma_f32_16x16x32_bf16(false, a0.v, false, b1.v, (short)0, acc01, false, false);
    acc10 = __builtin_amdgcn_wmma_f32_16x16x32_bf16(false, a1.v, false, b0.v, (short)0, acc10, false, false);
    acc11 = __builtin_amdgcn_wmma_f32_16x16x32_bf16(false, a1.v, false, b1.v, (short)0, acc11, false, false);
    __syncthreads();
  }

  const float* rm = rowmean + b * D_;
  float* cv = cov + (size_t)b * D_ * D_;
#pragma unroll
  for (int v = 0; v < 8; ++v) {
    const int r0w = row0 + wr * 32 + v + 8 * half;
    const int r1w = r0w + 16;
    const int c0w = col0 + wc * 32 + r;
    const int c1w = c0w + 16;
    cv[(size_t)r0w * D_ + c0w] = acc00[v] * (1.0f / (float)M_) - rm[r0w] * rm[c0w];
    cv[(size_t)r0w * D_ + c1w] = acc01[v] * (1.0f / (float)M_) - rm[r0w] * rm[c1w];
    cv[(size_t)r1w * D_ + c0w] = acc10[v] * (1.0f / (float)M_) - rm[r1w] * rm[c0w];
    cv[(size_t)r1w * D_ + c1w] = acc11[v] * (1.0f / (float)M_) - rm[r1w] * rm[c1w];
  }
}

// ---------------------------------------------------------------------------
// 5) Batched 256x256 GEMM:  D = alpha*A@op(B) + diag*I.
//    bf16 operands (shadow matrices) via double-buffered TDM panels; writes
//    fp32 result + bf16 shadow (conversion happens once, on the write side).
// ---------------------------------------------------------------------------
__global__ __launch_bounds__(256) void gemm256_kernel(const unsigned short* __restrict__ A,
                                                      const unsigned short* __restrict__ Bm,
                                                      float* __restrict__ Dm,
                                                      unsigned short* __restrict__ Dbf,
                                                      float alpha, float diag, int bT) {
  __shared__ unsigned short ldsA[2][128 * 32];   // 2 x 8 KB: [row][k]
  __shared__ unsigned short ldsB[2][64 * 32];    // 2 x 4 KB: NN [k][n], NT [n][k]

  const int b = blockIdx.y;
  const int wave = threadIdx.x >> 5;
  const int lane = threadIdx.x & 31;
  const int row0 = (blockIdx.x >> 2) * 128;
  const int col0 = (blockIdx.x & 3) * 64;
  const int wr = wave >> 1, wc = wave & 1;
  const int r = lane & 15, half = lane >> 4;

  const unsigned short* Ab = A  + (size_t)b * 65536;
  const unsigned short* Bb = Bm + (size_t)b * 65536;
  const unsigned offA[2] = { LDS_OFF(&ldsA[0][0]), LDS_OFF(&ldsA[1][0]) };
  const unsigned offB[2] = { LDS_OFF(&ldsB[0][0]), LDS_OFF(&ldsB[1][0]) };
  const int ra0 = wr * 32 + r, ra1 = ra0 + 16;
  const int cb0 = wc * 32 + r, cb1 = cb0 + 16;

  if (wave == 0) {
    tdm_load_2d(offA[0], Ab + (size_t)row0 * 256, 32, 32, 128, 256);
    if (bT) tdm_load_2d(offB[0], Bb + (size_t)col0 * 256, 32, 32, 64, 256);
    else    tdm_load_2d(offB[0], Bb + col0, 64, 64, 32, 256);
  }

  v8f acc00 = {}, acc01 = {}, acc10 = {}, acc11 = {};
  for (int s = 0; s < 8; ++s) {
    if (wave == 0) {
      if (s < 7) {
        const int kn = s * 32 + 32;
        const int nb = (s + 1) & 1;
        tdm_load_2d(offA[nb], Ab + (size_t)row0 * 256 + kn, 32, 32, 128, 256);
        if (bT) tdm_load_2d(offB[nb], Bb + (size_t)col0 * 256 + kn, 32, 32, 64, 256);
        else    tdm_load_2d(offB[nb], Bb + (size_t)kn * 256 + col0, 64, 64, 32, 256);
        __builtin_amdgcn_s_wait_tensorcnt(2);
      } else {
        __builtin_amdgcn_s_wait_tensorcnt(0);
      }
    }
    __syncthreads();
    const unsigned short* pA = ldsA[s & 1];
    const unsigned short* pB = ldsB[s & 1];

    BF16Frag a0, a1, b0, b1;
    a0.q[0] = *(const uint4*)(pA + ra0 * 32 + half * 8);
    a0.q[1] = *(const uint4*)(pA + ra0 * 32 + 16 + half * 8);
    a1.q[0] = *(const uint4*)(pA + ra1 * 32 + half * 8);
    a1.q[1] = *(const uint4*)(pA + ra1 * 32 + 16 + half * 8);
    if (bT) {
      b0.q[0] = *(const uint4*)(pB + cb0 * 32 + half * 8);
      b0.q[1] = *(const uint4*)(pB + cb0 * 32 + 16 + half * 8);
      b1.q[0] = *(const uint4*)(pB + cb1 * 32 + half * 8);
      b1.q[1] = *(const uint4*)(pB + cb1 * 32 + 16 + half * 8);
    } else {
#pragma unroll
      for (int e = 0; e < 16; ++e) {
        const int kb = (e < 8) ? half * 8 + e : 16 + half * 8 + (e - 8);
        b0.u[e] = pB[kb * 64 + cb0];
        b1.u[e] = pB[kb * 64 + cb1];
      }
    }
    acc00 = __builtin_amdgcn_wmma_f32_16x16x32_bf16(false, a0.v, false, b0.v, (short)0, acc00, false, false);
    acc01 = __builtin_amdgcn_wmma_f32_16x16x32_bf16(false, a0.v, false, b1.v, (short)0, acc01, false, false);
    acc10 = __builtin_amdgcn_wmma_f32_16x16x32_bf16(false, a1.v, false, b0.v, (short)0, acc10, false, false);
    acc11 = __builtin_amdgcn_wmma_f32_16x16x32_bf16(false, a1.v, false, b1.v, (short)0, acc11, false, false);
    __syncthreads();
  }

  float* Db = Dm + (size_t)b * 65536;
  unsigned short* Dbfb = Dbf + (size_t)b * 65536;
#pragma unroll
  for (int v = 0; v < 8; ++v) {
    const int r0w = row0 + wr * 32 + v + 8 * half;
    const int r1w = r0w + 16;
    const int c0w = col0 + wc * 32 + r;
    const int c1w = c0w + 16;
    float v00 = alpha * acc00[v]; if (r0w == c0w) v00 += diag;
    float v01 = alpha * acc01[v]; if (r0w == c1w) v01 += diag;
    float v10 = alpha * acc10[v]; if (r1w == c0w) v10 += diag;
    float v11 = alpha * acc11[v]; if (r1w == c1w) v11 += diag;
    Db[(size_t)r0w * 256 + c0w] = v00;
    Db[(size_t)r0w * 256 + c1w] = v01;
    Db[(size_t)r1w * 256 + c0w] = v10;
    Db[(size_t)r1w * 256 + c1w] = v11;
    Dbfb[(size_t)r0w * 256 + c0w] = f2bf(v00);
    Dbfb[(size_t)r0w * 256 + c1w] = f2bf(v01);
    Dbfb[(size_t)r1w * 256 + c0w] = f2bf(v10);
    Dbfb[(size_t)r1w * 256 + c1w] = f2bf(v11);
  }
}

// ---------------------------------------------------------------------------
// 6) trace of cov per batch -> 1/tr and sqrt(tr)
// ---------------------------------------------------------------------------
__global__ __launch_bounds__(256) void trace_kernel(const float* __restrict__ cov,
                                                    float* __restrict__ trinv,
                                                    float* __restrict__ trsqrt) {
  const int b = blockIdx.x;
  __shared__ float sh[256];
  sh[threadIdx.x] = cov[(size_t)b * 65536 + (size_t)threadIdx.x * 257];
  __syncthreads();
  for (int st = 128; st > 0; st >>= 1) {
    if (threadIdx.x < st) sh[threadIdx.x] += sh[threadIdx.x + st];
    __syncthreads();
  }
  if (threadIdx.x == 0) {
    const float tr = fmaxf(sh[0], 1e-12f);
    trinv[b]  = 1.0f / tr;
    trsqrt[b] = sqrtf(tr);
  }
}

// ---------------------------------------------------------------------------
// 7) elementwise:  D = c0 * (smode==1 ? scal[b] : 1) * X + diag*I
//    (writes fp32 result and bf16 shadow for downstream WMMA GEMMs)
// ---------------------------------------------------------------------------
__global__ __launch_bounds__(256) void ew_kernel(const float* __restrict__ X,
                                                 float* __restrict__ Dst,
                                                 unsigned short* __restrict__ Dbf,
                                                 const float* __restrict__ scal,
                                                 int smode, float c0, float diag) {
  const int idx = blockIdx.x * 256 + threadIdx.x;   // < 2,097,152
  const int b = idx >> 16;
  const int e = idx & 65535;
  const int i = e >> 8, j = e & 255;
  float f = c0;
  if (smode == 1) f *= scal[b];
  float v = X[idx] * f;
  if (i == j) v += diag;
  Dst[idx] = v;
  Dbf[idx] = f2bf(v);
}

// ---------------------------------------------------------------------------
// 8) Frobenius norm of att per batch
// ---------------------------------------------------------------------------
__global__ __launch_bounds__(256) void norm_kernel(const float* __restrict__ att,
                                                   float* __restrict__ nrm) {
  const int b = blockIdx.x;
  float s = 0.f;
  for (int i = threadIdx.x; i < 65536; i += 256) {
    const float v = att[(size_t)b * 65536 + i];
    s += v * v;
  }
  __shared__ float sh[256];
  sh[threadIdx.x] = s; __syncthreads();
  for (int st = 128; st > 0; st >>= 1) {
    if (threadIdx.x < st) sh[threadIdx.x] += sh[threadIdx.x + st];
    __syncthreads();
  }
  if (threadIdx.x == 0) nrm[b] = fmaxf(sqrtf(sh[0]), 1e-12f);
}

// ---------------------------------------------------------------------------
// 9) epilogue: y = out * (att / (att/nrm)) + out, extract upper triangle
// ---------------------------------------------------------------------------
__global__ __launch_bounds__(256) void final_kernel(const float* __restrict__ outm,
                                                    const float* __restrict__ att,
                                                    const float* __restrict__ nrm,
                                                    float* __restrict__ y) {
  const int b = blockIdx.x >> 8;
  const int i = blockIdx.x & 255;
  const int j = threadIdx.x;
  if (j < i) return;
  const size_t idx = (size_t)b * 65536 + (size_t)i * 256 + j;
  const float o = outm[idx];
  const float a = att[idx];
  const float n = nrm[b];
  const float normalized = a / n;
  const float ratio = a / normalized;
  const float val = o * ratio + o;
  const int pbase = i * 256 - (i * (i - 1)) / 2;
  y[(size_t)b * TRIU + pbase + (j - i)] = val;
}

// ---------------------------------------------------------------------------
// Host launcher
// ---------------------------------------------------------------------------
extern "C" void kernel_launch(void* const* d_in, const int* in_sizes, int n_in,
                              void* d_out, int out_size, void* d_ws, size_t ws_size,
                              hipStream_t stream) {
  (void)in_sizes; (void)n_in; (void)out_size; (void)ws_size;
  const float* x     = (const float*)d_in[0];
  const float* cw    = (const float*)d_in[1];
  const float* gamma = (const float*)d_in[2];
  const float* beta  = (const float*)d_in[3];
  float* out = (float*)d_out;

  char* base = (char*)d_ws;
  size_t off = 0;
  auto alloc = [&](size_t bytes) -> void* {
    void* q = base + off;
    off += (bytes + 255) & ~(size_t)255;
    return q;
  };

  unsigned short* Wb  = (unsigned short*)alloc((size_t)D_ * C_ * 2);
  unsigned short* xbf = (unsigned short*)alloc(XELEMS * 2);
  float*          z   = (float*)alloc(ZELEMS * 4);
  unsigned short* zbh = (unsigned short*)alloc(ZELEMS * 2);
  float* mu      = (float*)alloc(D_ * 4);
  float* istd    = (float*)alloc(D_ * 4);
  float* rowmean = (float*)alloc((size_t)B_ * D_ * 4);
  float* trinv   = (float*)alloc(B_ * 4);
  float* trsqrt  = (float*)alloc(B_ * 4);
  float* nrm     = (float*)alloc(B_ * 4);
  const size_t MATB = MATELEMS * 4;
  const size_t MATH = MATELEMS * 2;
  float* cov  = (float*)alloc(MATB);
  float* An   = (float*)alloc(MATB);
  float* Y1   = (float*)alloc(MATB);
  float* Z0   = (float*)alloc(MATB);
  float* Z1   = (float*)alloc(MATB);
  float* Tm   = (float*)alloc(MATB);
  float* S    = (float*)alloc(MATB);
  float* P0   = (float*)alloc(MATB);
  float* P1   = (float*)alloc(MATB);
  float* att  = (float*)alloc(MATB);
  float* outm = (float*)alloc(MATB);
  unsigned short* An_h   = (unsigned short*)alloc(MATH);
  unsigned short* Y1_h   = (unsigned short*)alloc(MATH);
  unsigned short* Z0_h   = (unsigned short*)alloc(MATH);
  unsigned short* Z1_h   = (unsigned short*)alloc(MATH);
  unsigned short* Tm_h   = (unsigned short*)alloc(MATH);
  unsigned short* S_h    = (unsigned short*)alloc(MATH);
  unsigned short* P0_h   = (unsigned short*)alloc(MATH);
  unsigned short* P1_h   = (unsigned short*)alloc(MATH);
  unsigned short* att_h  = (unsigned short*)alloc(MATH);
  unsigned short* outm_h = (unsigned short*)alloc(MATH);

  auto gemm = [&](const unsigned short* A, const unsigned short* Bm,
                  float* Dst, unsigned short* Dbf,
                  float alpha, float diag, int bT) {
    gemm256_kernel<<<dim3(8, B_), 256, 0, stream>>>(A, Bm, Dst, Dbf, alpha, diag, bT);
  };

  // conv + BN + ReLU + covpool
  cvt_w_kernel<<<2048, 256, 0, stream>>>(cw, Wb);
  cvt_x_kernel<<<50176, 256, 0, stream>>>(x, xbf);
  conv_gemm_kernel<<<dim3(13, 2, B_), 256, 0, stream>>>(Wb, xbf, z);
  bn_stats_kernel<<<D_, 256, 0, stream>>>(z, mu, istd);
  bn_relu_kernel<<<B_ * D_, 256, 0, stream>>>(z, mu, istd, gamma, beta, zbh, rowmean);
  cov_gemm_kernel<<<dim3(8, B_), 256, 0, stream>>>(zbh, rowmean, cov);
  trace_kernel<<<B_, 256, 0, stream>>>(cov, trinv, trsqrt);

  // sqrtm(cov) via trace-normalized Newton-Schulz (5 iters, all WMMA GEMMs)
  ew_kernel<<<8192, 256, 0, stream>>>(cov, An, An_h, trinv, 1, 1.0f, 0.0f);   // An = cov/tr
  ew_kernel<<<8192, 256, 0, stream>>>(cov, Z0, Z0_h, nullptr, 0, 0.0f, 1.0f); // Z0 = I
  float* Yb[2]  = {An, Y1};           unsigned short* Ybf[2]  = {An_h, Y1_h};
  float* Zb2[2] = {Z0, Z1};           unsigned short* Zbf[2]  = {Z0_h, Z1_h};
  int cy = 0, cz = 0;
  for (int it = 0; it < 5; ++it) {
    gemm(Zbf[cz], Ybf[cy], Tm, Tm_h, -0.5f, 1.5f, 0);               // T = 1.5I - 0.5 Z@Y
    gemm(Ybf[cy], Tm_h, Yb[cy ^ 1], Ybf[cy ^ 1], 1.0f, 0.0f, 0);    cy ^= 1; // Y' = Y@T
    gemm(Tm_h, Zbf[cz], Zb2[cz ^ 1], Zbf[cz ^ 1], 1.0f, 0.0f, 0);   cz ^= 1; // Z' = T@Z
  }
  ew_kernel<<<8192, 256, 0, stream>>>(Yb[cy], outm, outm_h, trsqrt, 1, 1.0f, 0.0f); // out = sqrt(tr)*Y

  // expm(-cov) via scaling (2^9) + order-6 Taylor (Horner) + 9 squarings
  ew_kernel<<<8192, 256, 0, stream>>>(cov, S, S_h, nullptr, 0, -1.0f / 512.0f, 0.0f);
  ew_kernel<<<8192, 256, 0, stream>>>(S, P0, P0_h, nullptr, 0, 1.0f / 6.0f, 1.0f); // P = I + S/6
  float* Pb[2] = {P0, P1};            unsigned short* Pbf[2] = {P0_h, P1_h};
  int cp = 0;
  for (int k = 5; k >= 1; --k) {
    gemm(S_h, Pbf[cp], Pb[cp ^ 1], Pbf[cp ^ 1], 1.0f / (float)k, 1.0f, 0); cp ^= 1;
  }
  for (int q = 0; q < 9; ++q) {
    gemm(Pbf[cp], Pbf[cp], Pb[cp ^ 1], Pbf[cp ^ 1], 1.0f, 0.0f, 0); cp ^= 1;
  }

  // attention epilogue
  gemm(outm_h, Pbf[cp], att, att_h, 1.0f, 0.0f, 1);   // att = out @ inv^T
  norm_kernel<<<B_, 256, 0, stream>>>(att, nrm);
  final_kernel<<<B_ * D_, 256, 0, stream>>>(outm, att, nrm, out);
}
